// Attention_Random_11192684774062
// MI455X (gfx1250) — matrix operand order repaired and checked by reading, not compiled, not run on hardware
//
#include <hip/hip_runtime.h>
#include <math.h>

// Fused attention for B=16, N=2048, D=256 (fp32 I/O, f16 WMMA compute, f32 accum).
// One block = one (batch, 16-query-row strip). 256 threads = 8 waves.
//   Pre-pass (if ws_size allows): K and V converted to f16 in d_ws.
//   Phase 1: S[16][2048] = (Q*scale) @ K^T via v_wmma_f32_16x16x32_f16 -> LDS (f32).
//            Explicit software pipeline: prefetch distance 2 chunks, cross-tile-pair
//            prefetch so WMMAs wait on partial loadcnt, never drain to 0.
//   Phase 2: row softmax in LDS (b128 scans); write p_attn f32 (coalesced);
//            pack normalized P as f16 IN PLACE into the low half of each score row.
//   Phase 3: out_t[d,n] = sum_m V[d,m]*P[n,m] via WMMA (M=d, N=q); rotating 2-slot
//            prefetch pipeline over m-chunks (V from global, P via ds_load_b128).
// LDS = 16 * 2052 * 4 B ~= 128 KB -> 2 blocks per 320 KB WGP (4 waves/SIMD).

typedef __attribute__((ext_vector_type(16))) _Float16 v16h;
typedef __attribute__((ext_vector_type(8)))  _Float16 v8h;
typedef __attribute__((ext_vector_type(4)))  _Float16 v4h;
typedef __attribute__((ext_vector_type(8)))  float    v8f;
typedef __attribute__((ext_vector_type(4)))  float    v4f;

#define BATCH 16
#define SEQ   2048
#define DK    256
#define SSTR  2052            // f32 score row stride: 2048 + 4 (conflict-free, 16B aligned)
#define HSTR  (2 * SSTR)      // same rows viewed as halfs

__device__ __forceinline__ v16h cvt4(v4f a, v4f b, v4f c, v4f d) {
  v16h r;
#pragma unroll
  for (int i = 0; i < 4; ++i) {
    r[i]      = (_Float16)a[i];
    r[i + 4]  = (_Float16)b[i];
    r[i + 8]  = (_Float16)c[i];
    r[i + 12] = (_Float16)d[i];
  }
  return r;
}

__device__ __forceinline__ v16h cat8(v8h lo, v8h hi) {
  v16h r;
#pragma unroll
  for (int i = 0; i < 8; ++i) { r[i] = lo[i]; r[i + 8] = hi[i]; }
  return r;
}

// Elementwise f32 -> f16 (4 elements/thread, b128 in / b64 out).
__global__ __launch_bounds__(256)
void cvt_f32_to_f16_kernel(const float* __restrict__ in, _Float16* __restrict__ out, int n4) {
  const int i = blockIdx.x * blockDim.x + threadIdx.x;
  if (i < n4) {
    v4f x = ((const v4f*)in)[i];
    v4h h;
#pragma unroll
    for (int j = 0; j < 4; ++j) h[j] = (_Float16)x[j];
    ((v4h*)out)[i] = h;
  }
}

template <bool F16IN>
__global__ __launch_bounds__(256)
void attn_fused_wmma_kernel(const float* __restrict__ Q,
                            const float* __restrict__ Kf, const _Float16* __restrict__ Kh,
                            const float* __restrict__ Vf, const _Float16* __restrict__ Vh,
                            float* __restrict__ Ot,    // [B, DK, SEQ]
                            float* __restrict__ P) {   // [B, SEQ, SEQ]
  extern __shared__ char smem[];
  float*    sS = (float*)smem;         // [16][SSTR] f32 scores; low half later = packed f16 P
  _Float16* sH = (_Float16*)smem;      // same storage viewed as halfs, row stride HSTR

  const int blk   = blockIdx.x;
  const int b     = blk >> 7;          // / (SEQ/16)
  const int qt    = blk & 127;
  const int qbase = qt << 4;
  const int tid   = threadIdx.x;
  const int wave  = tid >> 5;
  const int lane  = tid & 31;
  const int l16   = lane & 15;
  const int lhalf = lane >> 4;         // 0: lanes 0-15, 1: lanes 16-31

  const float scale = 0.0625f;         // 1/sqrt(256), exact in f16

  // ---------- Phase 1: S = (Q*scale) @ K^T ----------
  // A fragment (16x32 f16): lane holds row M=l16; k = kc+lhalf*16+{0..15}
  v16h aq[8];
  {
    const float* qrow = Q + (((size_t)b * SEQ + qbase) + l16) * DK;
#pragma unroll
    for (int c = 0; c < 8; ++c) {
      const int o1 = c * 32 + lhalf * 16;
      const int o2 = o1 + 8;
      v4f x0 = *(const v4f*)(qrow + o1);
      v4f x1 = *(const v4f*)(qrow + o1 + 4);
      v4f x2 = *(const v4f*)(qrow + o2);
      v4f x3 = *(const v4f*)(qrow + o2 + 4);
      x0 *= scale; x1 *= scale; x2 *= scale; x3 *= scale;
      aq[c] = cvt4(x0, x1, x2, x3);
    }
  }

  // Each wave owns 16 key tiles; two tiles (one pair) per j iteration.
  // B fragment (32x16 f16): lane holds column N=l16 (one key row),
  // k = kc + lhalf*16 + {0..15} contiguous in the key row (32B = 2x b128).
  if constexpr (F16IN) {
    const _Float16* Kbase = Kh + (size_t)b * SEQ * DK;
    const _Float16* kp0 = Kbase + (size_t)((wave << 4) * 16 + l16) * DK;
    const _Float16* kp1 = kp0 + (size_t)16 * DK;
    const int ol = lhalf * 16;
    // warm-up: chunks 0,1 of the first tile pair
    v16h nx0a = *(const v16h*)(kp0 + ol);
    v16h nx1a = *(const v16h*)(kp1 + ol);
    v16h nx0b = *(const v16h*)(kp0 + 32 + ol);
    v16h nx1b = *(const v16h*)(kp1 + 32 + ol);

    for (int j = 0; j < 8; ++j) {
      const size_t nstep = (j < 7) ? (size_t)32 * DK : 0;   // next pair (clamped at tail)
      const _Float16* kn0 = kp0 + nstep;
      const _Float16* kn1 = kp1 + nstep;
      v16h f0[8], f1[8];
      f0[0] = nx0a; f1[0] = nx1a; f0[1] = nx0b; f1[1] = nx1b;
      v8f acc0 = {}, acc1 = {};
#pragma unroll
      for (int c = 0; c < 8; ++c) {
        if (c < 6) {                       // prefetch chunk c+2 of current pair
          const int o = (c + 2) * 32 + ol;
          f0[c + 2] = *(const v16h*)(kp0 + o);
          f1[c + 2] = *(const v16h*)(kp1 + o);
        } else if (c == 6) {               // prefetch chunk 0 of next pair
          nx0a = *(const v16h*)(kn0 + ol);
          nx1a = *(const v16h*)(kn1 + ol);
        } else {                           // prefetch chunk 1 of next pair
          nx0b = *(const v16h*)(kn0 + 32 + ol);
          nx1b = *(const v16h*)(kn1 + 32 + ol);
        }
        acc0 = __builtin_amdgcn_wmma_f32_16x16x32_f16(false, aq[c], false, f0[c],
                                                      (short)0, acc0, false, false);
        acc1 = __builtin_amdgcn_wmma_f32_16x16x32_f16(false, aq[c], false, f1[c],
                                                      (short)0, acc1, false, false);
      }
      kp0 = kn0; kp1 = kn1;
      const int col0 = ((wave << 4) + 2 * j) * 16 + l16;   // C: VGPR v -> row v+8*lhalf, col l16
#pragma unroll
      for (int v = 0; v < 8; ++v) {
        sS[(size_t)(v + 8 * lhalf) * SSTR + col0]      = acc0[v];
        sS[(size_t)(v + 8 * lhalf) * SSTR + col0 + 16] = acc1[v];
      }
    }
  } else {
    for (int j = 0; j < 8; ++j) {
      const int ct0 = (wave << 4) + 2 * j;
      const size_t koff0 = (size_t)b * SEQ * DK + (size_t)(ct0 * 16 + l16) * DK;
      const size_t koff1 = koff0 + (size_t)16 * DK;
      v8f acc0 = {}, acc1 = {};
#pragma unroll
      for (int c = 0; c < 8; ++c) {
        const int o = c * 32 + lhalf * 16;
        const float* k0 = Kf + koff0 + o;
        const float* k1 = Kf + koff1 + o;
        v16h bk0 = cvt4(*(const v4f*)(k0), *(const v4f*)(k0 + 4),
                        *(const v4f*)(k0 + 8), *(const v4f*)(k0 + 12));
        v16h bk1 = cvt4(*(const v4f*)(k1), *(const v4f*)(k1 + 4),
                        *(const v4f*)(k1 + 8), *(const v4f*)(k1 + 12));
        acc0 = __builtin_amdgcn_wmma_f32_16x16x32_f16(false, aq[c], false, bk0,
                                                      (short)0, acc0, false, false);
        acc1 = __builtin_amdgcn_wmma_f32_16x16x32_f16(false, aq[c], false, bk1,
                                                      (short)0, acc1, false, false);
      }
      const int col0 = ct0 * 16 + l16;
#pragma unroll
      for (int v = 0; v < 8; ++v) {
        sS[(size_t)(v + 8 * lhalf) * SSTR + col0]      = acc0[v];
        sS[(size_t)(v + 8 * lhalf) * SSTR + col0 + 16] = acc1[v];
      }
    }
  }
  __syncthreads();

  // ---------- Phase 2: row softmax (wave w handles rows 2w, 2w+1), vectorized ----------
#pragma unroll
  for (int rr = 0; rr < 2; ++rr) {
    const int r = wave * 2 + rr;
    const float* row = sS + (size_t)r * SSTR;
    float m = -3.0e38f;
    for (int i = lane * 4; i < SEQ; i += 128) {          // b128 LDS reads, conflict-free
      v4f x = *(const v4f*)(row + i);
      m = fmaxf(fmaxf(fmaxf(m, x[0]), x[1]), fmaxf(x[2], x[3]));
    }
#pragma unroll
    for (int off = 16; off > 0; off >>= 1) m = fmaxf(m, __shfl_xor(m, off, 32));
    float s = 0.0f;
    for (int i = lane * 4; i < SEQ; i += 128) {
      v4f x = *(const v4f*)(row + i);
      s += __expf(x[0] - m) + __expf(x[1] - m) + __expf(x[2] - m) + __expf(x[3] - m);
    }
#pragma unroll
    for (int off = 16; off > 0; off >>= 1) s += __shfl_xor(s, off, 32);
    const float inv = 1.0f / s;
    float*    prow = P + ((size_t)b * SEQ + qbase + r) * SEQ;
    _Float16* hrow = sH + (size_t)r * HSTR;
    // In-place pack: writes at iter k cover f32 slots [64k, 64k+63]; reads at iter k' > k
    // start at slot 128k' > 64k+63, and same-iter writes are data-dependent on the read.
    for (int i = lane * 4; i < SEQ; i += 128) {
      v4f x = *(const v4f*)(row + i);
      v4f p;
#pragma unroll
      for (int e = 0; e < 4; ++e) p[e] = __expf(x[e] - m) * inv;
      *(v4f*)(prow + i) = p;                             // coalesced 16B f32 stores of p_attn
      v4h hv;
#pragma unroll
      for (int e = 0; e < 4; ++e) hv[e] = (_Float16)p[e];
      *(v4h*)(hrow + i) = hv;                            // ds_store_b64, packed f16 P
    }
  }
  __syncthreads();

  // ---------- Phase 3: Ot[d, n] = sum_m V[d,m] * P[n,m]  (M=d, N=q, K=m) ----------
  // Wave w owns d-tiles at d0 = 32w and d0+16. B fragment: lane holds col N=l16 (query q),
  // m = mc + lhalf*16 + {0..15} from packed f16 P (2x ds_load_b128).
  v8f o0 = {}, o1 = {};
  const int d0 = wave * 32;
  const size_t vbase    = (size_t)b * DK * SEQ;
  const size_t vrow0off = vbase + (size_t)(d0 + l16) * SEQ;
  const size_t vrow1off = vrow0off + (size_t)16 * SEQ;
  const _Float16* sProw = sH + (size_t)l16 * HSTR;

  if constexpr (F16IN) {
    const _Float16* v0p = Vh + vrow0off;
    const _Float16* v1p = Vh + vrow1off;
    // rotating 2-slot pipeline, prefetch distance 64 m-elements (2 chunks)
    v16h bp[2]; v8h al0[2], ah0[2], al1[2], ah1[2];
#pragma unroll
    for (int s = 0; s < 2; ++s) {
      const int mc = s * 32;
      bp[s]  = *(const v16h*)(sProw + mc + lhalf * 16);
      const int oa = mc + lhalf * 16, obp = oa + 8;
      al0[s] = *(const v8h*)(v0p + oa); ah0[s] = *(const v8h*)(v0p + obp);
      al1[s] = *(const v8h*)(v1p + oa); ah1[s] = *(const v8h*)(v1p + obp);
    }
    for (int mc = 0; mc < SEQ; mc += 64) {
#pragma unroll
      for (int s = 0; s < 2; ++s) {
        const int cur = mc + s * 32;
        int pf = cur + 64; if (pf >= SEQ) pf = cur;      // clamped tail prefetch, uniform flow
        v16h bpc = bp[s];
        v8h l0 = al0[s], h0 = ah0[s], l1 = al1[s], h1 = ah1[s];
        bp[s]  = *(const v16h*)(sProw + pf + lhalf * 16);
        const int oa = pf + lhalf * 16, obp = oa + 8;
        al0[s] = *(const v8h*)(v0p + oa); ah0[s] = *(const v8h*)(v0p + obp);
        al1[s] = *(const v8h*)(v1p + oa); ah1[s] = *(const v8h*)(v1p + obp);
        o0 = __builtin_amdgcn_wmma_f32_16x16x32_f16(false, cat8(l0, h0), false, bpc,
                                                    (short)0, o0, false, false);
        o1 = __builtin_amdgcn_wmma_f32_16x16x32_f16(false, cat8(l1, h1), false, bpc,
                                                    (short)0, o1, false, false);
      }
    }
  } else {
    const float* vr0 = Vf + vrow0off;
    const float* vr1 = Vf + vrow1off;
    for (int mc = 0; mc < SEQ; mc += 32) {
      v16h bpc = *(const v16h*)(sProw + mc + lhalf * 16);
      const int oa = mc + lhalf * 16, obp = oa + 8;
      v16h av0 = cvt4(*(const v4f*)(vr0 + oa), *(const v4f*)(vr0 + oa + 4),
                      *(const v4f*)(vr0 + obp), *(const v4f*)(vr0 + obp + 4));
      v16h av1 = cvt4(*(const v4f*)(vr1 + oa), *(const v4f*)(vr1 + oa + 4),
                      *(const v4f*)(vr1 + obp), *(const v4f*)(vr1 + obp + 4));
      o0 = __builtin_amdgcn_wmma_f32_16x16x32_f16(false, av0, false, bpc, (short)0, o0, false, false);
      o1 = __builtin_amdgcn_wmma_f32_16x16x32_f16(false, av1, false, bpc, (short)0, o1, false, false);
    }
  }

  // D layout: VGPR v -> d = d_tile + v + 8*lhalf, n = qbase + l16 (contiguous across lanes)
  float* outb = Ot + (size_t)b * DK * SEQ + qbase + l16;
#pragma unroll
  for (int v = 0; v < 8; ++v) {
    const int d = d0 + v + 8 * lhalf;
    outb[(size_t)d * SEQ]        = o0[v];
    outb[(size_t)(d + 16) * SEQ] = o1[v];
  }
}

extern "C" void kernel_launch(void* const* d_in, const int* in_sizes, int n_in,
                              void* d_out, int out_size, void* d_ws, size_t ws_size,
                              hipStream_t stream) {
  const float* Q = (const float*)d_in[0];   // [16, 2048, 256]
  const float* K = (const float*)d_in[1];   // [16, 2048, 256]
  const float* V = (const float*)d_in[2];   // [16, 256, 2048]
  // d_in[3] = mask flag (0) -> ignored (deterministic reference has no dropout)

  float* out_t = (float*)d_out;                        // [16, 256, 2048]
  float* P     = out_t + (size_t)BATCH * DK * SEQ;     // [16, 2048, 2048]

  const dim3 grid(BATCH * (SEQ / 16));   // 2048 blocks
  const dim3 block(256);                 // 8 waves
  const size_t shmem = (size_t)16 * SSTR * sizeof(float);   // ~128 KB -> 2 blocks/WGP

  const size_t nElem = (size_t)BATCH * SEQ * DK;            // 8,388,608 per tensor
  const size_t need  = 2 * nElem * sizeof(_Float16);        // f16 K + f16 V

  if (ws_size >= need) {
    _Float16* Kh = (_Float16*)d_ws;
    _Float16* Vh = Kh + nElem;
    const int n4 = (int)(nElem / 4);
    cvt_f32_to_f16_kernel<<<n4 / 256, 256, 0, stream>>>(K, Kh, n4);
    cvt_f32_to_f16_kernel<<<n4 / 256, 256, 0, stream>>>(V, Vh, n4);
    attn_fused_wmma_kernel<true><<<grid, block, shmem, stream>>>(Q, K, Kh, V, Vh, out_t, P);
  } else {
    attn_fused_wmma_kernel<false><<<grid, block, shmem, stream>>>(Q, K, nullptr, V, nullptr, out_t, P);
  }
}
